// SelfAttention_6485400617384
// MI455X (gfx1250) — matrix operand order repaired and checked by reading, not compiled, not run on hardware
//
#include <hip/hip_runtime.h>

// ---------------------------------------------------------------------------
// CDNA5 (gfx1250) wave32 WMMA self-attention pipeline, bf16 matrix path.
// ---------------------------------------------------------------------------

typedef __attribute__((ext_vector_type(16))) __bf16 v16bf;
typedef __attribute__((ext_vector_type(8)))  __bf16 v8bf;
typedef __attribute__((ext_vector_type(8)))  float  v8f;

#define B_DIM   2
#define S_LEN   2048
#define E_DIM   1024
#define HID     1024
#define HEADS   16
#define DH      64            // head dim
#define QKV_W   (3 * HID)     // 3072
#define QKV_STR 192           // per-head qkv row stride (3*HID/HEADS)
#define ROWS    (B_DIM * S_LEN)  // 4096
#define SCALE_F 0.022097086912079608f  // (2*E)^-0.5 = 1/sqrt(2048)

// ---- WMMA wrapper ----------------------------------------------------------
__device__ __forceinline__ v8f wmma_bf16(v16bf a, v16bf b, v8f c) {
  // 8 args: (neg_a, A, neg_b, B, c_mod, C, reuse_a, reuse_b)
  return __builtin_amdgcn_wmma_f32_16x16x32_bf16(false, a, false, b, (short)0, c,
                                                 false, false);
}

// A-fragment (16x32 bf16): lane sel=lane>>4 picks contiguous K
// [2*base0, 2*base0+16) with base0 = sel*8 (matches frag_b).
__device__ __forceinline__ v16bf frag_a(const __bf16* row, int base0) {
  const v8bf lo = *(const v8bf*)(row + 2 * base0);
  const v8bf hi = *(const v8bf*)(row + 2 * base0 + 8);
  v16bf f;
#pragma unroll
  for (int j = 0; j < 8; ++j) { f[j] = lo[j]; f[j + 8] = hi[j]; }
  return f;
}

// B-fragment (32x16 bf16): lane holds K = sel*16 .. sel*16+15 of one column,
// contiguous when B is stored column-major ([N][K]) -> single 32B load.
__device__ __forceinline__ v16bf frag_b(const __bf16* colK) {
  return *(const v16bf*)(colK);
}

// ---- 1) LayerNorm -> bf16 --------------------------------------------------
__global__ __launch_bounds__(256)
void layernorm_bf16_kernel(const float* __restrict__ x,
                           const float* __restrict__ g,
                           const float* __restrict__ b,
                           __bf16* __restrict__ xn) {
  __shared__ float red[256];
  const int row = blockIdx.x;
  const int tid = threadIdx.x;
  const float4 v = ((const float4*)(x + (size_t)row * E_DIM))[tid];

  red[tid] = v.x + v.y + v.z + v.w;
  __syncthreads();
#pragma unroll
  for (int o = 128; o > 0; o >>= 1) {
    if (tid < o) red[tid] += red[tid + o];
    __syncthreads();
  }
  const float mu = red[0] * (1.0f / E_DIM);
  __syncthreads();

  const float dx = v.x - mu, dy = v.y - mu, dz = v.z - mu, dw = v.w - mu;
  red[tid] = dx * dx + dy * dy + dz * dz + dw * dw;
  __syncthreads();
#pragma unroll
  for (int o = 128; o > 0; o >>= 1) {
    if (tid < o) red[tid] += red[tid + o];
    __syncthreads();
  }
  const float rs = rsqrtf(red[0] * (1.0f / E_DIM) + 1e-5f);

  const float4 gg = ((const float4*)g)[tid];
  const float4 bb = ((const float4*)b)[tid];
  __bf16* orow = xn + (size_t)row * E_DIM + tid * 4;
  orow[0] = (__bf16)(dx * rs * gg.x + bb.x);
  orow[1] = (__bf16)(dy * rs * gg.y + bb.y);
  orow[2] = (__bf16)(dz * rs * gg.z + bb.z);
  orow[3] = (__bf16)(dw * rs * gg.w + bb.w);
}

// ---- 2) weight transpose + cvt: out[n][k] = bf16(in[k][n]) -----------------
__global__ __launch_bounds__(256)
void transpose_cvt_kernel(const float* __restrict__ in, __bf16* __restrict__ out,
                          int K, int N) {
  const size_t idx = (size_t)blockIdx.x * 256 + threadIdx.x;
  const int n = (int)(idx / K);
  const int k = (int)(idx % K);
  out[idx] = (__bf16)in[(size_t)k * N + n];
}

// ---- 3) generic bf16 WMMA GEMM: C[M,N] = A[M,K] * Bt[N,K]^T + bias ---------
template <typename OutT>
__global__ __launch_bounds__(256)
void gemm_bf16_kernel(const __bf16* __restrict__ A, const __bf16* __restrict__ Bt,
                      const float* __restrict__ bias, OutT* __restrict__ C,
                      int M, int N, int K) {
  const int lane  = threadIdx.x & 31;
  const int wave  = threadIdx.x >> 5;
  const int sel   = lane >> 4;
  const int l16   = lane & 15;
  const int base0 = sel * 8;

  const int wm = wave >> 2;                     // 0..1
  const int wn = wave & 3;                      // 0..3
  const int mW = blockIdx.y * 64 + wm * 32;     // wave tile 32x32
  const int nW = blockIdx.x * 128 + wn * 32;

  const __bf16* a0p = A  + (size_t)(mW + l16)      * K;
  const __bf16* a1p = A  + (size_t)(mW + 16 + l16) * K;
  const __bf16* b0p = Bt + (size_t)(nW + l16)      * K + sel * 16;
  const __bf16* b1p = Bt + (size_t)(nW + 16 + l16) * K + sel * 16;

  v8f c00 = {}, c01 = {}, c10 = {}, c11 = {};
  for (int k0 = 0; k0 < K; k0 += 32) {
    __builtin_prefetch(a0p + k0 + 128, 0, 1);   // global_prefetch_b8
    __builtin_prefetch(b0p + k0 + 128, 0, 1);
    const v16bf a0 = frag_a(a0p + k0, base0);
    const v16bf a1 = frag_a(a1p + k0, base0);
    const v16bf b0 = frag_b(b0p + k0);
    const v16bf b1 = frag_b(b1p + k0);
    c00 = wmma_bf16(a0, b0, c00);
    c01 = wmma_bf16(a0, b1, c01);
    c10 = wmma_bf16(a1, b0, c10);
    c11 = wmma_bf16(a1, b1, c11);
  }

  const float bv0 = bias[nW + l16];
  const float bv1 = bias[nW + 16 + l16];
#pragma unroll
  for (int r = 0; r < 8; ++r) {
    const int row0 = mW + r + 8 * sel;
    const int row1 = row0 + 16;
    C[(size_t)row0 * N + nW + l16]      = (OutT)(c00[r] + bv0);
    C[(size_t)row0 * N + nW + 16 + l16] = (OutT)(c01[r] + bv1);
    C[(size_t)row1 * N + nW + l16]      = (OutT)(c10[r] + bv0);
    C[(size_t)row1 * N + nW + 16 + l16] = (OutT)(c11[r] + bv1);
  }
}

// ---- 4) flash attention over one (head, batch) per block -------------------
// qkv (bf16) viewed as [HEADS*B][S][192]; q = +0, k = +64, v = +128.
// Output written bf16 at flat [H][B][S][64] (reinterpreted by GEMM2).
__global__ __launch_bounds__(256)
void attention_kernel(const __bf16* __restrict__ qkv, __bf16* __restrict__ attOut) {
  __shared__ __align__(16) __bf16 Kb[32][64];       // [key][d]
  __shared__ __align__(16) __bf16 Vt[64][32];       // [d][key] (transposed)
  __shared__ __align__(16) __bf16 Pb[8][16][32];    // per-wave P relayout patch

  const int tid   = threadIdx.x;
  const int lane  = tid & 31;
  const int wave  = tid >> 5;
  const int sel   = lane >> 4;
  const int l16   = lane & 15;
  const int base0 = sel * 8;

  const int hb    = blockIdx.x;                     // h*B + b  (0..31)
  const int qBase = blockIdx.y * 128 + wave * 16;   // 16 queries per wave
  const __bf16* blk = qkv + (size_t)hb * S_LEN * QKV_STR;

  // Q fragments for d=0..31 and d=32..63 (rows qBase + l16)
  const __bf16* qrow = blk + (size_t)(qBase + l16) * QKV_STR;
  const v16bf qf0 = frag_a(qrow, base0);
  const v16bf qf1 = frag_a(qrow + 32, base0);

  v8f o0 = {}, o1 = {}, o2 = {}, o3 = {};
  float mrun[8], lrun[8];
#pragma unroll
  for (int r = 0; r < 8; ++r) { mrun[r] = -3.0e38f; lrun[r] = 0.0f; }

  for (int kb = 0; kb < S_LEN; kb += 32) {
    __syncthreads();
    // Cooperative stage of 32-key K/V tiles (8 contiguous elems per thread).
    {
      const int idx = tid * 8;
      const int kk  = idx >> 6;
      const int dd  = idx & 63;
      const __bf16* krow = blk + (size_t)(kb + kk) * QKV_STR + 64 + dd;
      const __bf16* vrow = blk + (size_t)(kb + kk) * QKV_STR + 128 + dd;
#pragma unroll
      for (int e = 0; e < 8; ++e) {
        Kb[kk][dd + e] = krow[e];
        Vt[dd + e][kk] = vrow[e];          // transpose for contiguous B-frags
      }
    }
    __syncthreads();

    // scores: S = Q (16x64) * K^T (64x32keys), two 16-key tiles
    v8f s1 = {}, s2 = {};
    s1 = wmma_bf16(qf0, frag_b(&Kb[l16][sel * 16]),      s1);
    s1 = wmma_bf16(qf1, frag_b(&Kb[l16][32 + sel * 16]), s1);
    s2 = wmma_bf16(qf0, frag_b(&Kb[16 + l16][sel * 16]),      s2);
    s2 = wmma_bf16(qf1, frag_b(&Kb[16 + l16][32 + sel * 16]), s2);

    // online softmax (row = r + 8*sel, keys spread across 16 lanes)
    float p1[8], p2[8];
#pragma unroll
    for (int r = 0; r < 8; ++r) {
      const float v1 = s1[r] * SCALE_F;
      const float v2 = s2[r] * SCALE_F;
      float tmax = fmaxf(v1, v2);
#pragma unroll
      for (int off = 1; off < 16; off <<= 1)
        tmax = fmaxf(tmax, __shfl_xor(tmax, off, 32));
      const float newm  = fmaxf(mrun[r], tmax);
      const float alpha = __expf(mrun[r] - newm);
      const float e1 = __expf(v1 - newm);
      const float e2 = __expf(v2 - newm);
      float tsum = e1 + e2;
#pragma unroll
      for (int off = 1; off < 16; off <<= 1)
        tsum += __shfl_xor(tsum, off, 32);
      lrun[r] = lrun[r] * alpha + tsum;
      mrun[r] = newm;
      p1[r] = e1; p2[r] = e2;
      o0[r] *= alpha; o1[r] *= alpha; o2[r] *= alpha; o3[r] *= alpha;
    }

    // Re-layout P (C-format) -> A-format through per-wave LDS patch.
#pragma unroll
    for (int r = 0; r < 8; ++r) {
      Pb[wave][r + 8 * sel][l16]      = (__bf16)p1[r];
      Pb[wave][r + 8 * sel][16 + l16] = (__bf16)p2[r];
    }
    const v16bf pf = frag_a(&Pb[wave][l16][0], base0);   // wave-local, DS in-order

    // O += P (16x32) * V (32keys x 64)
    o0 = wmma_bf16(pf, frag_b(&Vt[l16][sel * 16]),      o0);
    o1 = wmma_bf16(pf, frag_b(&Vt[16 + l16][sel * 16]), o1);
    o2 = wmma_bf16(pf, frag_b(&Vt[32 + l16][sel * 16]), o2);
    o3 = wmma_bf16(pf, frag_b(&Vt[48 + l16][sel * 16]), o3);
  }

  // epilogue: divide by softmax denominator, store bf16 at flat [H][B][S][64]
  __bf16* outBlk = attOut + (size_t)hb * S_LEN * DH;
#pragma unroll
  for (int r = 0; r < 8; ++r) {
    const float inv = 1.0f / lrun[r];
    __bf16* orow = outBlk + (size_t)(qBase + r + 8 * sel) * DH;
    orow[l16]      = (__bf16)(o0[r] * inv);
    orow[16 + l16] = (__bf16)(o1[r] * inv);
    orow[32 + l16] = (__bf16)(o2[r] * inv);
    orow[48 + l16] = (__bf16)(o3[r] * inv);
  }
}

// ---------------------------------------------------------------------------
extern "C" void kernel_launch(void* const* d_in, const int* in_sizes, int n_in,
                              void* d_out, int out_size, void* d_ws, size_t ws_size,
                              hipStream_t stream) {
  const float* x    = (const float*)d_in[0];
  const float* ln_g = (const float*)d_in[1];
  const float* ln_b = (const float*)d_in[2];
  const float* W1   = (const float*)d_in[3];
  const float* b1   = (const float*)d_in[4];
  const float* W2   = (const float*)d_in[5];
  const float* b2   = (const float*)d_in[6];
  float* out = (float*)d_out;

  char* ws = (char*)d_ws;
  size_t off = 0;
  auto alloc = [&](size_t bytes) -> void* {
    void* p = ws + off;
    off = (off + bytes + 255) & ~(size_t)255;
    return p;
  };
  __bf16* xn   = (__bf16*)alloc((size_t)ROWS * E_DIM * 2);    //  8 MB
  __bf16* w1t  = (__bf16*)alloc((size_t)QKV_W * E_DIM * 2);   //  6 MB
  __bf16* w2t  = (__bf16*)alloc((size_t)HID * HID * 2);       //  2 MB
  __bf16* qkvb = (__bf16*)alloc((size_t)ROWS * QKV_W * 2);    // 24 MB
  __bf16* att  = (__bf16*)alloc((size_t)ROWS * HID * 2);      //  8 MB

  // 1) LayerNorm -> bf16 activations
  layernorm_bf16_kernel<<<ROWS, 256, 0, stream>>>(x, ln_g, ln_b, xn);
  // 2) weights -> transposed bf16
  transpose_cvt_kernel<<<(QKV_W * E_DIM) / 256, 256, 0, stream>>>(W1, w1t, E_DIM, QKV_W);
  transpose_cvt_kernel<<<(HID * HID) / 256, 256, 0, stream>>>(W2, w2t, HID, HID);
  // 3) QKV projection (WMMA), bf16 staging output
  gemm_bf16_kernel<__bf16><<<dim3(QKV_W / 128, ROWS / 64), 256, 0, stream>>>(
      xn, w1t, b1, qkvb, ROWS, QKV_W, E_DIM);
  // 4) flash attention per (head, batch)
  attention_kernel<<<dim3(HEADS * B_DIM, S_LEN / 128), 256, 0, stream>>>(qkvb, att);
  // 5) output projection (WMMA), f32 output
  gemm_bf16_kernel<float><<<dim3(HID / 128, ROWS / 64), 256, 0, stream>>>(
      att, w2t, b2, out, ROWS, HID, E_DIM);
}
